// PatchedLlama3Attention_52398601011490
// MI455X (gfx1250) — compile-verified, matrix operand order checked by reading, not run on hardware
//
#include <hip/hip_runtime.h>

// ---------------- problem constants ----------------
#define S_LEN 2048
#define HID   4096
#define NH    32
#define NKV   8
#define HD    128
// GROUPS = NH/NKV = 4

typedef __bf16 bf16_t;
typedef bf16_t bf16x16 __attribute__((ext_vector_type(16)));
typedef float  floatx8 __attribute__((ext_vector_type(8)));

union FragB16 {
  bf16x16 v;
  uint4   q[2];
  unsigned u[8];
};

__device__ __forceinline__ unsigned short f32_to_bf16(float f) {
  union { float f; unsigned u; } x; x.f = f;
  unsigned r = x.u + 0x7FFFu + ((x.u >> 16) & 1u);
  return (unsigned short)(r >> 16);
}

__device__ __forceinline__ unsigned bf16pack(float a, float b) {
  union { float f; unsigned u; } x, y; x.f = a; y.f = b;
  unsigned ra = x.u + 0x7FFFu + ((x.u >> 16) & 1u);
  unsigned rb = y.u + 0x7FFFu + ((y.u >> 16) & 1u);
  return (ra >> 16) | (rb & 0xFFFF0000u);
}

__device__ __forceinline__ FragB16 load_frag(const unsigned short* __restrict__ row, int off) {
  FragB16 f;
  f.q[0] = *(const uint4*)(row + off);
  f.q[1] = *(const uint4*)(row + off + 16);
  return f;
}

__device__ __forceinline__ floatx8 wmma_bf16(bf16x16 a, bf16x16 b, floatx8 c) {
  return __builtin_amdgcn_wmma_f32_16x16x32_bf16(false, a, false, b, (short)0, c, false, false);
}

// ---- async global->LDS staging (ASYNCcnt path), with portable fallback ----
typedef int v4i_vs __attribute__((vector_size(16)));

#if defined(__gfx1250__) && __has_builtin(__builtin_amdgcn_global_load_async_to_lds_b128)
#define HAVE_ASYNC_LDS 1
__device__ __forceinline__ void cp16_g2l(const unsigned short* g, unsigned short* l) {
  __builtin_amdgcn_global_load_async_to_lds_b128(
      (__attribute__((address_space(1))) v4i_vs*)g,
      (__attribute__((address_space(3))) v4i_vs*)l, 0, 0);
}
#else
#define HAVE_ASYNC_LDS 0
__device__ __forceinline__ void cp16_g2l(const unsigned short* g, unsigned short* l) {
  *(uint4*)l = *(const uint4*)g;   // sync fallback: global_load_b128 + ds_store_b128
}
#endif

#if __has_builtin(__builtin_amdgcn_s_wait_asynccnt)
#define WAIT_ASYNC(n) __builtin_amdgcn_s_wait_asynccnt(n)
#else
#define WAIT_ASYNC(n) asm volatile("s_wait_asynccnt %0" :: "i"(n) : "memory")
#endif

// ---------------- elementwise converts ----------------
__global__ __launch_bounds__(256) void cvt_bf16_kernel(const float* __restrict__ in,
                                                       unsigned short* __restrict__ out,
                                                       long long n) {
  long long i = ((long long)blockIdx.x * blockDim.x + threadIdx.x) * 8;
  if (i >= n) return;
  union { uint4 u4; unsigned u[4]; } o;
#pragma unroll
  for (int t = 0; t < 4; ++t) o.u[t] = bf16pack(in[i + 2*t], in[i + 2*t + 1]);
  *(uint4*)(out + i) = o.u4;
}

// RoPE on Q: Qf32[s,4096] -> Qbf[h][s][d]
__global__ __launch_bounds__(256) void rope_q_kernel(const float* __restrict__ Qf,
                                                     const float* __restrict__ cosb,
                                                     const float* __restrict__ sinb,
                                                     unsigned short* __restrict__ Qbf) {
  int idx = blockIdx.x * blockDim.x + threadIdx.x;   // S*NH*(HD/2)
  if (idx >= S_LEN * NH * (HD/2)) return;
  int j = idx & 63; int t = idx >> 6; int h = t & (NH-1); int s = t / NH;
  const float* r = Qf + (size_t)s * HID + h * HD;
  float q1 = r[2*j], q2 = r[2*j + 1];
  float c = cosb[s*64 + j], sn = sinb[s*64 + j];
  unsigned short* o = Qbf + ((size_t)h * S_LEN + s) * HD;
  o[j]      = f32_to_bf16(q1 * c - q2 * sn);
  o[64 + j] = f32_to_bf16(q2 * c + q1 * sn);
}

// RoPE on K: Kf32[s,1024] -> Kbf[kv][s][d]
__global__ __launch_bounds__(256) void rope_k_kernel(const float* __restrict__ Kf,
                                                     const float* __restrict__ cosb,
                                                     const float* __restrict__ sinb,
                                                     unsigned short* __restrict__ Kbf) {
  int idx = blockIdx.x * blockDim.x + threadIdx.x;   // S*NKV*(HD/2)
  if (idx >= S_LEN * NKV * (HD/2)) return;
  int j = idx & 63; int t = idx >> 6; int h = t & (NKV-1); int s = t / NKV;
  const float* r = Kf + (size_t)s * (NKV*HD) + h * HD;
  float k1 = r[2*j], k2 = r[2*j + 1];
  float c = cosb[s*64 + j], sn = sinb[s*64 + j];
  unsigned short* o = Kbf + ((size_t)h * S_LEN + s) * HD;
  o[j]      = f32_to_bf16(k1 * c - k2 * sn);
  o[64 + j] = f32_to_bf16(k2 * c + k1 * sn);
}

// V transpose: Vf32[s,1024] -> Vt[kv][d][s]
__global__ __launch_bounds__(256) void vt_kernel(const float* __restrict__ Vf,
                                                 unsigned short* __restrict__ Vt) {
  int idx = blockIdx.x * blockDim.x + threadIdx.x;   // S*NKV*HD
  if (idx >= S_LEN * NKV * HD) return;
  int o = idx & (NKV*HD - 1); int s = idx >> 10;
  int h = o >> 7; int d = o & (HD-1);
  Vt[((size_t)h * HD + d) * S_LEN + s] = f32_to_bf16(Vf[idx]);
}

// ---------------- GEMM: C[M,N] = A[M,K](bf16) * B[N,K](bf16)^T, f32 out ----
// block = 256 threads (8 waves). Block tile 128M x 64N. A/B tiles double-
// buffered in LDS via async global->LDS copies (ASYNCcnt), padded stride 40
// elements (80B) for conflict-free ds_load_b128 fragment reads.
#define LPAD 40

__global__ __launch_bounds__(256) void gemm_bf16_nt(const unsigned short* __restrict__ A,
                                                    const unsigned short* __restrict__ B,
                                                    float* __restrict__ C,
                                                    int M, int N, int K) {
  __shared__ __align__(16) unsigned short lA[2][128 * LPAD];
  __shared__ __align__(16) unsigned short lB[2][64 * LPAD];

  const int tid  = threadIdx.x;
  const int lane = tid & 31;
  const int wave = tid >> 5;
  const int col  = lane & 15;
  const int half = lane >> 4;
  const int m0 = blockIdx.y * 128;
  const int n0 = blockIdx.x * 64;

  // staging coordinates: each thread moves three 16B chunks per K-step
  const int srow = tid >> 2;            // 0..63
  const int sc   = (tid & 3) * 8;       // element offset of chunk in row
  const unsigned short* gA0 = A + (size_t)(m0 + srow) * K + sc;
  const unsigned short* gA1 = A + (size_t)(m0 + srow + 64) * K + sc;
  const unsigned short* gB  = B + (size_t)(n0 + srow) * K + sc;
  const int lro  = srow * LPAD + sc;

  // prologue: stage k0 = 0 into buffer 0
  cp16_g2l(gA0, &lA[0][lro]);
  cp16_g2l(gA1, &lA[0][lro + 64 * LPAD]);
  cp16_g2l(gB,  &lB[0][lro]);

  floatx8 acc0 = {}, acc1 = {}, acc2 = {}, acc3 = {};
  const int fa = (wave * 16 + col) * LPAD + half * 8;
  const int fb = col * LPAD + half * 8;

  for (int k0 = 0; k0 < K; k0 += 32) {
    const int buf = (k0 >> 5) & 1;
    if (k0 + 32 < K) {
      const int nxt = buf ^ 1;
      cp16_g2l(gA0 + k0 + 32, &lA[nxt][lro]);
      cp16_g2l(gA1 + k0 + 32, &lA[nxt][lro + 64 * LPAD]);
      cp16_g2l(gB  + k0 + 32, &lB[nxt][lro]);
      WAIT_ASYNC(3);                 // current buffer's 3 copies done
    } else {
      WAIT_ASYNC(0);                 // final tile fully arrived
    }
    __syncthreads();

    FragB16 a, b0, b1, b2, b3;
    const unsigned short* pa = &lA[buf][fa];
    a.q[0]  = *(const uint4*)pa;
    a.q[1]  = *(const uint4*)(pa + 16);
    const unsigned short* pb = &lB[buf][fb];
    b0.q[0] = *(const uint4*)pb;
    b0.q[1] = *(const uint4*)(pb + 16);
    b1.q[0] = *(const uint4*)(pb + 16 * LPAD);
    b1.q[1] = *(const uint4*)(pb + 16 * LPAD + 16);
    b2.q[0] = *(const uint4*)(pb + 32 * LPAD);
    b2.q[1] = *(const uint4*)(pb + 32 * LPAD + 16);
    b3.q[0] = *(const uint4*)(pb + 48 * LPAD);
    b3.q[1] = *(const uint4*)(pb + 48 * LPAD + 16);

    acc0 = wmma_bf16(a.v, b0.v, acc0);
    acc1 = wmma_bf16(a.v, b1.v, acc1);
    acc2 = wmma_bf16(a.v, b2.v, acc2);
    acc3 = wmma_bf16(a.v, b3.v, acc3);

    __syncthreads();                 // protect buf from next stage overwrite
  }

#pragma unroll
  for (int v = 0; v < 8; ++v) {
    int m = m0 + wave * 16 + v + 8 * half;
    float* crow = C + (size_t)m * N + n0 + col;
    crow[0]  = acc0[v];
    crow[16] = acc1[v];
    crow[32] = acc2[v];
    crow[48] = acc3[v];
  }
}

// ---------------- flash attention (transposed-score formulation) ----------
// grid = (S/16/8, NH), block = 256 (8 waves); wave -> one 16-query tile.
// Sᵀ = K·Qᵀ  (lane owns query column col, key rows v+8*half)
// Oᵀ += Vᵀ·Pᵀ (C fragment of Sᵀ -> bf16 pack == B fragment of Pᵀ, no shuffles)
__global__ __launch_bounds__(256) void attn_kernel(const unsigned short* __restrict__ Qbf,
                                                   const unsigned short* __restrict__ Kbf,
                                                   const unsigned short* __restrict__ Vt,
                                                   unsigned short* __restrict__ ctx) {
  const int lane = threadIdx.x & 31;
  const int wave = threadIdx.x >> 5;
  const int col  = lane & 15;
  const int half = lane >> 4;
  const int h  = blockIdx.y;
  const int qt = blockIdx.x * 8 + wave;
  const int q0 = qt * 16;
  const int q  = q0 + col;
  const int kvh = h >> 2;                 // GROUPS = 4
  const float scale = 0.08838834764831845f;  // 1/sqrt(128)

  const unsigned short* qrow = Qbf + ((size_t)h * S_LEN + q) * HD;
  FragB16 qf[4];
#pragma unroll
  for (int kb = 0; kb < 4; ++kb) qf[kb] = load_frag(qrow, kb * 32 + half * 8);

  const unsigned short* kbase = Kbf + (size_t)kvh * S_LEN * HD;
  const unsigned short* vbase = Vt  + (size_t)kvh * HD * S_LEN;

  floatx8 acc[8];
#pragma unroll
  for (int df = 0; df < 8; ++df) acc[df] = (floatx8){};
  float m = -1e30f, l = 0.0f;

  const int kend = q0 + 16;               // causal bound
  for (int k0 = 0; k0 < kend; k0 += 32) {
    floatx8 s0 = {}, s1 = {};
    const unsigned short* krow0 = kbase + (size_t)(k0 + col) * HD;
    const unsigned short* krow1 = krow0 + 16 * HD;
    __builtin_prefetch(krow0 + 32 * HD, 0, 3);   // next key tile
    __builtin_prefetch(krow1 + 32 * HD, 0, 3);
#pragma unroll
    for (int kb = 0; kb < 4; ++kb) {
      FragB16 ka = load_frag(krow0, kb * 32 + half * 8);
      s0 = wmma_bf16(ka.v, qf[kb].v, s0);
    }
#pragma unroll
    for (int kb = 0; kb < 4; ++kb) {
      FragB16 ka = load_frag(krow1, kb * 32 + half * 8);
      s1 = wmma_bf16(ka.v, qf[kb].v, s1);
    }
    // ---- scale + causal mask + online softmax (two shuffles) ----
    float f0[8], f1[8];
    float pm = -1e30f;
    const int keyb = k0 + 8 * half;
#pragma unroll
    for (int v = 0; v < 8; ++v) {
      int key0 = keyb + v;
      int key1 = key0 + 16;
      f0[v] = s0[v] * scale + (key0 > q ? -1e9f : 0.0f);
      f1[v] = s1[v] * scale + (key1 > q ? -1e9f : 0.0f);
      pm = fmaxf(pm, fmaxf(f0[v], f1[v]));
    }
    pm = fmaxf(pm, __shfl_xor(pm, 16, 32));
    float mnew  = fmaxf(m, pm);
    float alpha = __expf(m - mnew);
    float rs = 0.0f;
    float p0[8], p1[8];
#pragma unroll
    for (int v = 0; v < 8; ++v) {
      p0[v] = __expf(f0[v] - mnew);
      p1[v] = __expf(f1[v] - mnew);
      rs += p0[v] + p1[v];
    }
    rs += __shfl_xor(rs, 16, 32);
    l = l * alpha + rs;
    m = mnew;
    // ---- pack P^T B-fragment (C layout == B layout, lane-local) ----
    FragB16 pf;
#pragma unroll
    for (int i = 0; i < 4; ++i) pf.u[i]     = bf16pack(p0[2*i], p0[2*i + 1]);
#pragma unroll
    for (int i = 0; i < 4; ++i) pf.u[4 + i] = bf16pack(p1[2*i], p1[2*i + 1]);
    // ---- O^T += V^T * P^T : 8 WMMA over the 128 output dims ----
#pragma unroll
    for (int df = 0; df < 8; ++df) {
      const unsigned short* vrow = vbase + (size_t)(df * 16 + col) * S_LEN;
      FragB16 vf = load_frag(vrow, k0 + half * 8);
      acc[df] = acc[df] * alpha;
      acc[df] = wmma_bf16(vf.v, pf.v, acc[df]);
    }
  }

  // ---- normalize and store ctx[s][h*128+d] (bf16) ----
  const float inv = 1.0f / l;
  unsigned short* crow = ctx + (size_t)q * HID + h * HD;
#pragma unroll
  for (int df = 0; df < 8; ++df) {
    union { uint4 u4; unsigned u[4]; } o;
#pragma unroll
    for (int i = 0; i < 4; ++i)
      o.u[i] = bf16pack(acc[df][2*i] * inv, acc[df][2*i + 1] * inv);
    *(uint4*)(crow + df * 16 + 8 * half) = o.u4;
  }
}

// ---------------- workspace layout (bytes) ----------------
#define OFF_XBF  ((size_t)0)            // 16 MB  X bf16 [s][4096]
#define OFF_WBF  ((size_t)16777216)     // 32 MB  weight bf16 (reused per GEMM)
#define OFF_QF   ((size_t)50331648)     // 32 MB  Q f32 [s][4096]; later reused as ctx bf16
#define OFF_KF   ((size_t)83886080)     //  8 MB  K f32 [s][1024]
#define OFF_VF   ((size_t)92274688)     //  8 MB  V f32 [s][1024]
#define OFF_QBF  ((size_t)100663296)    // 16 MB  Q bf16 [h][s][128]
#define OFF_KBF  ((size_t)117440512)    //  4 MB  K bf16 [kv][s][128]
#define OFF_VT   ((size_t)121634816)    //  4 MB  V^T bf16 [kv][128][s]
// total 120 MB

extern "C" void kernel_launch(void* const* d_in, const int* in_sizes, int n_in,
                              void* d_out, int out_size, void* d_ws, size_t ws_size,
                              hipStream_t stream) {
  (void)in_sizes; (void)n_in; (void)out_size; (void)ws_size;
  const float* X    = (const float*)d_in[0];
  const float* cosb = (const float*)d_in[3];
  const float* sinb = (const float*)d_in[4];
  const float* Wq   = (const float*)d_in[5];
  const float* Wk   = (const float*)d_in[6];
  const float* Wv   = (const float*)d_in[7];
  const float* Wo   = (const float*)d_in[8];
  char* ws = (char*)d_ws;
  unsigned short* Xbf = (unsigned short*)(ws + OFF_XBF);
  unsigned short* Wbf = (unsigned short*)(ws + OFF_WBF);
  float*          Qf  = (float*)(ws + OFF_QF);
  float*          Kf  = (float*)(ws + OFF_KF);
  float*          Vf  = (float*)(ws + OFF_VF);
  unsigned short* Qbf = (unsigned short*)(ws + OFF_QBF);
  unsigned short* Kbf = (unsigned short*)(ws + OFF_KBF);
  unsigned short* Vt  = (unsigned short*)(ws + OFF_VT);
  unsigned short* ctx = (unsigned short*)(ws + OFF_QF);  // reuse (Qf dead after rope)
  float* out = (float*)d_out;

  const long long nX  = (long long)S_LEN * HID;       // 8388608
  const long long nWq = (long long)HID * HID;         // 16777216
  const long long nWk = (long long)(NKV*HD) * HID;    // 4194304

  // X -> bf16
  cvt_bf16_kernel<<<(unsigned)(nX/8/256),  256, 0, stream>>>(X,  Xbf, nX);

  // Q = X * Wq^T
  cvt_bf16_kernel<<<(unsigned)(nWq/8/256), 256, 0, stream>>>(Wq, Wbf, nWq);
  gemm_bf16_nt<<<dim3(HID/64, S_LEN/128), 256, 0, stream>>>(Xbf, Wbf, Qf, S_LEN, HID, HID);
  // K = X * Wk^T
  cvt_bf16_kernel<<<(unsigned)(nWk/8/256), 256, 0, stream>>>(Wk, Wbf, nWk);
  gemm_bf16_nt<<<dim3((NKV*HD)/64, S_LEN/128), 256, 0, stream>>>(Xbf, Wbf, Kf, S_LEN, NKV*HD, HID);
  // V = X * Wv^T
  cvt_bf16_kernel<<<(unsigned)(nWk/8/256), 256, 0, stream>>>(Wv, Wbf, nWk);
  gemm_bf16_nt<<<dim3((NKV*HD)/64, S_LEN/128), 256, 0, stream>>>(Xbf, Wbf, Vf, S_LEN, NKV*HD, HID);

  // RoPE + layout transforms
  rope_q_kernel<<<(S_LEN*NH*(HD/2))/256, 256, 0, stream>>>(Qf, cosb, sinb, Qbf);
  rope_k_kernel<<<(S_LEN*NKV*(HD/2))/256, 256, 0, stream>>>(Kf, cosb, sinb, Kbf);
  vt_kernel<<<(S_LEN*NKV*HD)/256, 256, 0, stream>>>(Vf, Vt);

  // flash attention -> ctx bf16 [s][4096]
  attn_kernel<<<dim3(S_LEN/16/8, NH), 256, 0, stream>>>(Qbf, Kbf, Vt, ctx);

  // out = ctx * Wo^T  (f32 output)
  cvt_bf16_kernel<<<(unsigned)(nWq/8/256), 256, 0, stream>>>(Wo, Wbf, nWq);
  gemm_bf16_nt<<<dim3(HID/64, S_LEN/128), 256, 0, stream>>>(ctx, Wbf, out, S_LEN, HID, HID);
}
